// GNNModel_3332894622673
// MI455X (gfx1250) — matrix operand order repaired and checked by reading, not compiled, not run on hardware
//
#include <hip/hip_runtime.h>
#include <cstdint>
#include <cstddef>

// ---------- types ----------
typedef __attribute__((ext_vector_type(16))) __bf16        v16bf;
typedef __attribute__((ext_vector_type(8)))  float         v8f;
typedef __attribute__((ext_vector_type(4)))  unsigned int  v4u;
typedef __attribute__((ext_vector_type(8)))  unsigned int  v8u;

#define C_IN  256
#define C_HID 256
#define C_OUT 64
#define KDIM  256   // K for both GEMMs
#define KT    8     // KDIM / 32

__device__ __forceinline__ unsigned short f2bf(float f) {
  unsigned int u = __float_as_uint(f);
  u += 0x7FFFu + ((u >> 16) & 1u);   // round-to-nearest-even
  return (unsigned short)(u >> 16);
}
__device__ __forceinline__ float bf2f(unsigned int h) {
  return __uint_as_float(h << 16);
}
__device__ __forceinline__ void atomAddF(float* p, float v) {
  unsafeAtomicAdd(p, v);             // native global_atomic_add_f32
}

// ---------- conversion / packing ----------
__global__ void k_f2bf(const float* __restrict__ in, unsigned short* __restrict__ out, int n) {
  int t = blockIdx.x * blockDim.x + threadIdx.x;
  if (t < n) out[t] = f2bf(in[t]);
}

// Pack row-major W[K,N] f32 into WMMA B-fragment order (bf16):
// fragment (kt,nt) = 512 bf16, per lane 16 contiguous; lane<16 -> N col = nt*16+lane,
// element j -> K = kt*32 + (j&7) + (j>=8 ? 16 : 0) ; lanes 16..31 add 8.
__global__ void k_packW(const float* __restrict__ W, unsigned short* __restrict__ out,
                        int K, int N) {
  int t = blockIdx.x * blockDim.x + threadIdx.x;
  if (t >= K * N) return;
  int frag = t >> 9;
  int r    = t & 511;
  int lane = r >> 4, j = r & 15;
  int NT = N >> 4;
  int kt = frag / NT, nt = frag - kt * NT;
  int k = kt * 32 + ((lane >= 16) ? 8 : 0) + (j & 7) + ((j >= 8) ? 16 : 0);
  int n = nt * 16 + (lane & 15);
  out[t] = f2bf(W[k * N + n]);
}

// ---------- degree / normalization ----------
__global__ void k_fill1(float* __restrict__ p, int n) {
  int t = blockIdx.x * blockDim.x + threadIdx.x;
  if (t < n) p[t] = 1.0f;                       // self-loop
}
__global__ void k_deg_edges(const int* __restrict__ dst, float* __restrict__ deg, int nE) {
  int t = blockIdx.x * blockDim.x + threadIdx.x;
  if (t < nE) atomAddF(&deg[dst[t]], 1.0f);
}
__global__ void k_rsqrt(float* __restrict__ p, int n) {
  int t = blockIdx.x * blockDim.x + threadIdx.x;
  if (t < n) p[t] = rsqrtf(p[t]);               // deg >= 1 always
}

// ---------- WMMA GEMM: C[M,N](bf16) = A[M,256](bf16) x Bpacked(LDS) ----------
template <int N>
__global__ __launch_bounds__(256) void k_gemm_bf16(
    const unsigned short* __restrict__ A,    // [Mtiles*16, 256] row-major bf16
    const unsigned short* __restrict__ Bp,   // packed fragments (KT * N/16 * 512 bf16)
    unsigned short* __restrict__ C,          // [M, N] bf16
    int Mtiles) {
  constexpr int NT = N / 16;
  __shared__ unsigned short ldsB[KT * NT * 512];

  // Cooperative stage of packed W into LDS (before any early return).
  {
    const v4u* gsrc = (const v4u*)Bp;
    v4u* ldst = (v4u*)ldsB;
    constexpr int NVEC = KT * NT * 512 * 2 / 16;   // 16B vectors
    for (int i = threadIdx.x; i < NVEC; i += 256) ldst[i] = gsrc[i];
  }
  __syncthreads();

  int wave = (int)((blockIdx.x * blockDim.x + threadIdx.x) >> 5);
  int lane = threadIdx.x & 31;
  if (wave >= Mtiles) return;                 // wave-uniform: EXEC stays all-ones
  int l  = lane & 15;
  int hi = lane >> 4;

  // A-fragment per ISA 16-bit A layout: lane<16 holds row M=l, K {0..7,16..23};
  // lane>=16 holds row M=l, K {8..15,24..31}  (all +kt*32).
  const unsigned short* ap = A + (size_t)(wave * 16 + l) * KDIM + hi * 8;
  v16bf a[KT];
#pragma unroll
  for (int kt = 0; kt < KT; ++kt) {
    v4u lo = *(const v4u*)(ap + kt * 32);
    v4u h4 = *(const v4u*)(ap + kt * 32 + 16);
    v8u w = {lo[0], lo[1], lo[2], lo[3], h4[0], h4[1], h4[2], h4[3]};
    a[kt] = __builtin_bit_cast(v16bf, w);
  }

  const unsigned short* lb = ldsB + lane * 16;
#pragma unroll
  for (int nt = 0; nt < NT; nt += 2) {
    v8f c0 = {}, c1 = {};
#pragma unroll
    for (int kt = 0; kt < KT; ++kt) {
      v8u w0 = *(const v8u*)(lb + (kt * NT + nt) * 512);
      v8u w1 = *(const v8u*)(lb + (kt * NT + nt + 1) * 512);
      v16bf b0 = __builtin_bit_cast(v16bf, w0);
      v16bf b1 = __builtin_bit_cast(v16bf, w1);
      c0 = __builtin_amdgcn_wmma_f32_16x16x32_bf16(
          false, a[kt], false, b0, (short)0, c0, false, false);
      c1 = __builtin_amdgcn_wmma_f32_16x16x32_bf16(
          false, a[kt], false, b1, (short)0, c1, false, false);
    }
    // C/D layout: element r -> row = wave*16 + hi*8 + r, col = nt*16 + l
    unsigned short* cp0 = C + (size_t)(wave * 16 + hi * 8) * N + nt * 16 + l;
#pragma unroll
    for (int r = 0; r < 8; ++r) {
      cp0[(size_t)r * N]      = f2bf(c0[r]);
      cp0[(size_t)r * N + 16] = f2bf(c1[r]);
    }
  }
}

// ---------- self-loop init (also fully initializes accumulators) ----------
__global__ void k_self1(const unsigned short* __restrict__ hbf, const float* __restrict__ dinv,
                        float* __restrict__ acc, int n) {
  int t = blockIdx.x * blockDim.x + threadIdx.x;
  if (t >= n) return;
  int i = t >> 8;                               // F = 256
  float di = dinv[i];
  acc[t] = bf2f(hbf[t]) * di * di;
}
__global__ void k_self2(const unsigned short* __restrict__ hbf, const float* __restrict__ dinv,
                        const float* __restrict__ bias, float* __restrict__ out, int n) {
  int t = blockIdx.x * blockDim.x + threadIdx.x;
  if (t >= n) return;
  int i = t >> 6, f = t & 63;                   // F = 64
  float di = dinv[i];
  out[t] = bf2f(hbf[t]) * di * di + bias[f];
}

__global__ void k_relu_bias(const float* __restrict__ acc, const float* __restrict__ bias,
                            unsigned short* __restrict__ outbf, int n) {
  int t = blockIdx.x * blockDim.x + threadIdx.x;
  if (t >= n) return;
  int f = t & 255;
  float v = acc[t] + bias[f];
  outbf[t] = f2bf(v > 0.f ? v : 0.f);
}

// ---------- edge scatter: one wave per edge, L2-resident atomics ----------
template <int F>
__global__ __launch_bounds__(256) void k_edges(
    const unsigned short* __restrict__ hbf,
    const int* __restrict__ src, const int* __restrict__ dst,
    const float* __restrict__ dinv, float* __restrict__ acc, int nE) {
  int wave = (int)((blockIdx.x * blockDim.x + threadIdx.x) >> 5);
  if (wave >= nE) return;
  int lane = threadIdx.x & 31;
  int s = src[wave], d = dst[wave];
  float nrm = dinv[s] * dinv[d];
  if constexpr (F == 256) {
    v4u v = *(const v4u*)(hbf + (size_t)s * F + lane * 8);   // 8 bf16 / lane
    float* ap = acc + (size_t)d * F + lane * 8;
#pragma unroll
    for (int j = 0; j < 4; ++j) {
      unsigned int w = v[j];
      atomAddF(ap + 2 * j,     bf2f(w & 0xFFFFu) * nrm);
      atomAddF(ap + 2 * j + 1, bf2f(w >> 16)     * nrm);
    }
  } else {  // F == 64: 2 bf16 / lane
    unsigned int w = *(const unsigned int*)(hbf + (size_t)s * F + lane * 2);
    float* ap = acc + (size_t)d * F + lane * 2;
    atomAddF(ap,     bf2f(w & 0xFFFFu) * nrm);
    atomAddF(ap + 1, bf2f(w >> 16)     * nrm);
  }
}

// ---------- launch ----------
extern "C" void kernel_launch(void* const* d_in, const int* in_sizes, int n_in,
                              void* d_out, int out_size, void* d_ws, size_t ws_size,
                              hipStream_t stream) {
  const float* x   = (const float*)d_in[0];
  const int*   ei  = (const int*)d_in[1];
  const float* W1  = (const float*)d_in[2];
  const float* b1  = (const float*)d_in[3];
  const float* W2  = (const float*)d_in[4];
  const float* b2  = (const float*)d_in[5];
  float*       out = (float*)d_out;

  const int nN = in_sizes[0] / C_IN;     // 50000
  const int nE = in_sizes[1] / 2;        // 1,000,000
  const int* srcp = ei;
  const int* dstp = ei + nE;

  char* ws = (char*)d_ws;
  size_t off = 0;
  auto carve = [&](size_t bytes) -> void* {
    void* p = ws + off;
    off = (off + bytes + 255) & ~(size_t)255;
    return p;
  };
  float*          dinv = (float*)carve((size_t)nN * 4);
  unsigned short* W1p  = (unsigned short*)carve((size_t)C_IN * C_HID * 2);
  unsigned short* W2p  = (unsigned short*)carve((size_t)C_HID * C_OUT * 2);
  unsigned short* xbf  = (unsigned short*)carve((size_t)nN * C_IN * 2);   // reused as act bf16
  unsigned short* hbf  = (unsigned short*)carve((size_t)nN * C_HID * 2);  // reused as h2 bf16
  float*          acc  = (float*)carve((size_t)nN * C_HID * 4);
  (void)ws_size; (void)n_in; (void)out_size;

  const int B = 256;
  auto cdiv = [](long long a, long long b) { return (int)((a + b - 1) / b); };

  const long long nx = (long long)nN * C_IN;    // 12.8M
  const long long nh = (long long)nN * C_HID;   // 12.8M
  const long long no = (long long)nN * C_OUT;   // 3.2M
  const int Mtiles = nN / 16;                    // 3125

  // prep
  k_f2bf <<<cdiv(nx, B), B, 0, stream>>>(x, xbf, (int)nx);
  k_packW<<<cdiv(C_IN * C_HID, B), B, 0, stream>>>(W1, W1p, C_IN, C_HID);
  k_packW<<<cdiv(C_HID * C_OUT, B), B, 0, stream>>>(W2, W2p, C_HID, C_OUT);
  k_fill1<<<cdiv(nN, B), B, 0, stream>>>(dinv, nN);
  k_deg_edges<<<cdiv(nE, B), B, 0, stream>>>(dstp, dinv, nE);
  k_rsqrt<<<cdiv(nN, B), B, 0, stream>>>(dinv, nN);

  // layer 1
  k_gemm_bf16<C_HID><<<cdiv(Mtiles, 8), B, 0, stream>>>(xbf, W1p, hbf, Mtiles);
  k_self1<<<cdiv(nh, B), B, 0, stream>>>(hbf, dinv, acc, (int)nh);
  k_edges<256><<<cdiv((long long)nE * 32, B), B, 0, stream>>>(hbf, srcp, dstp, dinv, acc, nE);
  k_relu_bias<<<cdiv(nh, B), B, 0, stream>>>(acc, b1, xbf, (int)nh);   // xbf := act bf16

  // layer 2
  k_gemm_bf16<C_OUT><<<cdiv(Mtiles, 8), B, 0, stream>>>(xbf, W2p, hbf, Mtiles); // hbf := h2
  k_self2<<<cdiv(no, B), B, 0, stream>>>(hbf, dinv, b2, out, (int)no);
  k_edges<64><<<cdiv((long long)nE * 32, B), B, 0, stream>>>(hbf, srcp, dstp, dinv, out, nE);
}
